// GNN_3066606649898
// MI455X (gfx1250) — matrix operand order repaired and checked by reading, not compiled, not run on hardware
//
#include <hip/hip_runtime.h>
#include <stdint.h>

// ---------------------------------------------------------------------------
// Types for CDNA5 WMMA (gfx1250, wave32)
// ---------------------------------------------------------------------------
typedef __attribute__((ext_vector_type(16))) __bf16 v16bf;
typedef __attribute__((ext_vector_type(8)))  float  v8f;

__device__ __forceinline__ __bf16 f2bf(float f) {
  // round-to-nearest-even fp32 -> bf16
  unsigned u = __builtin_bit_cast(unsigned, f);
  unsigned r = u + 0x7FFFu + ((u >> 16) & 1u);
  unsigned short s = (unsigned short)(r >> 16);
  return __builtin_bit_cast(__bf16, s);
}

__device__ __forceinline__ v16bf ldfrag(const __bf16* p0, const __bf16* p1) {
  union { uint4 u[2]; v16bf v; } t;
  t.u[0] = *(const uint4*)p0;
  t.u[1] = *(const uint4*)p1;
  return t.v;
}

// Async 16B copy global -> LDS (CDNA5 GLOBAL_LOAD_ASYNC_TO_LDS_B128, ASYNCcnt).
// lds_off = LDS byte offset (low 32 bits of flat shared address).
__device__ __forceinline__ void async_cp16(unsigned lds_off, const __bf16* gptr) {
  asm volatile("global_load_async_to_lds_b128 %0, %1, off"
               :: "v"(lds_off), "v"(gptr)
               : "memory");
}
__device__ __forceinline__ void wait_async0() {
  asm volatile("s_wait_asynccnt 0x0" ::: "memory");
}
__device__ __forceinline__ unsigned lds_off32(const void* shared_ptr) {
  return (unsigned)(uintptr_t)shared_ptr;  // addrspace(3) offset in low 32 bits
}

// ---------------------------------------------------------------------------
// bf16 WMMA GEMM:  C[M,Nc] = A[M,K] @ B[K,Nc] + bias[Nc]
// A given as bf16 row-major; B given pre-transposed bf16 row-major [Nc,K].
// Block tile 128x128, 256 threads = 8 waves; each wave computes 32x64 via
// 2x4 v_wmma_f32_16x16x32_bf16 accumulators. K stepped by 32.
// Tiles are streamed with async global->LDS copies, double-buffered, one
// barrier per K-step (own-wave s_wait_asynccnt 0 precedes the barrier).
// Fragment layouts follow CDNA5 ISA 7.12.2 (wave32).
// ---------------------------------------------------------------------------
__global__ __launch_bounds__(256)
void gemm_bf16_wmma(const __bf16* __restrict__ A, const __bf16* __restrict__ Bt,
                    const float* __restrict__ bias, float* __restrict__ C,
                    int M, int Nc, int K) {
  __shared__ __bf16 As[2][128 * 32];
  __shared__ __bf16 Bs[2][128 * 32];
  const int tid  = threadIdx.x;
  const int lane = tid & 31;
  const int w    = tid >> 5;
  const int mBase = blockIdx.y * 128, nBase = blockIdx.x * 128;
  const int wm = (w & 3) * 32;   // 4 waves along M
  const int wn = (w >> 2) * 64;  // 2 waves along N
  const int l15 = lane & 15, lh = lane >> 4;

  // Per-thread copy slots: 2 x 16B for As, 2 x 16B for Bs per K-step.
  const int lin0 = tid * 8, lin1 = 2048 + tid * 8;   // half-elem index in 128x32
  const int rowA = lin0 >> 5, colA = lin0 & 31;
  const int rowB = lin1 >> 5, colB = lin1 & 31;
  const __bf16* gA0 = A  + (size_t)(mBase + rowA) * K + colA;
  const __bf16* gA1 = A  + (size_t)(mBase + rowB) * K + colB;
  const __bf16* gB0 = Bt + (size_t)(nBase + rowA) * K + colA;
  const __bf16* gB1 = Bt + (size_t)(nBase + rowB) * K + colB;
  const unsigned oA0[2] = { lds_off32(&As[0][rowA * 32 + colA]),
                            lds_off32(&As[1][rowA * 32 + colA]) };
  const unsigned oA1[2] = { lds_off32(&As[0][rowB * 32 + colB]),
                            lds_off32(&As[1][rowB * 32 + colB]) };
  const unsigned oB0[2] = { lds_off32(&Bs[0][rowA * 32 + colA]),
                            lds_off32(&Bs[1][rowA * 32 + colA]) };
  const unsigned oB1[2] = { lds_off32(&Bs[0][rowB * 32 + colB]),
                            lds_off32(&Bs[1][rowB * 32 + colB]) };

  v8f acc[2][4];
#pragma unroll
  for (int nf = 0; nf < 4; ++nf) {
    float bv = bias[nBase + wn + nf * 16 + l15];
#pragma unroll
    for (int mf = 0; mf < 2; ++mf)
#pragma unroll
      for (int r = 0; r < 8; ++r) acc[mf][nf][r] = bv;
  }

  // Prologue: stream first K-slice into buffer 0.
  async_cp16(oA0[0], gA0);
  async_cp16(oA1[0], gA1);
  async_cp16(oB0[0], gB0);
  async_cp16(oB1[0], gB1);

  for (int k0 = 0, it = 0; k0 < K; k0 += 32, ++it) {
    const int cur = it & 1;
    wait_async0();        // my async writes are in LDS
    __syncthreads();      // everyone's are visible
    if (k0 + 32 < K) {    // stream next slice into the other buffer
      const int nxt = cur ^ 1;
      async_cp16(oA0[nxt], gA0 + k0 + 32);
      async_cp16(oA1[nxt], gA1 + k0 + 32);
      async_cp16(oB0[nxt], gB0 + k0 + 32);
      async_cp16(oB1[nxt], gB1 + k0 + 32);
    }

    v16bf a[2], b[4];
#pragma unroll
    for (int mf = 0; mf < 2; ++mf) {
      // A 16x32 bf16 fragment: lane holds row (lane&15); K chunks
      // [8*lh .. 8*lh+7] and [16+8*lh .. 16+8*lh+7]  (ISA 7.12.2)
      const __bf16* pa = &As[cur][(wm + mf * 16 + l15) * 32 + lh * 8];
      a[mf] = ldfrag(pa, pa + 16);
    }
#pragma unroll
    for (int nf = 0; nf < 4; ++nf) {
      // B 32x16 bf16 fragment: lane holds col (lane&15); K chunks
      // [8*lh .. 8*lh+7] and [16+8*lh .. 16+8*lh+7]
      const __bf16* pb = &Bs[cur][(wn + nf * 16 + l15) * 32 + lh * 8];
      b[nf] = ldfrag(pb, pb + 16);
    }
#pragma unroll
    for (int mf = 0; mf < 2; ++mf)
#pragma unroll
      for (int nf = 0; nf < 4; ++nf)
        acc[mf][nf] = __builtin_amdgcn_wmma_f32_16x16x32_bf16(
            false, a[mf], false, b[nf], (short)0, acc[mf][nf], false, false);
  }

#pragma unroll
  for (int mf = 0; mf < 2; ++mf)
#pragma unroll
    for (int nf = 0; nf < 4; ++nf)
#pragma unroll
      for (int r = 0; r < 8; ++r) {
        int m = mBase + wm + mf * 16 + r + lh * 8;  // C/D layout ISA 7.12.2
        int n = nBase + wn + nf * 16 + l15;
        C[(size_t)m * Nc + n] = acc[mf][nf][r];
      }
}

// ---------------------------------------------------------------------------
// Small utility kernels
// ---------------------------------------------------------------------------
__global__ __launch_bounds__(256)
void cvt_f32_bf16(const float* __restrict__ in, __bf16* __restrict__ out, int n4) {
  int i = blockIdx.x * blockDim.x + threadIdx.x;
  if (i >= n4) return;
  float4 v = ((const float4*)in)[i];
  union { __bf16 h[4]; uint2 u; } t;
  t.h[0] = f2bf(v.x); t.h[1] = f2bf(v.y); t.h[2] = f2bf(v.z); t.h[3] = f2bf(v.w);
  ((uint2*)out)[i] = t.u;
}

__global__ __launch_bounds__(256)
void transpose_cvt(const float* __restrict__ W, __bf16* __restrict__ Wt,
                   int K, int Nc) {
  int i = blockIdx.x * blockDim.x + threadIdx.x;
  if (i >= K * Nc) return;
  int k = i / Nc, n = i - k * Nc;
  Wt[(size_t)n * K + k] = f2bf(W[i]);
}

__global__ __launch_bounds__(256)
void fill_f32(float* p, float v, int n) {
  int i = blockIdx.x * blockDim.x + threadIdx.x;
  if (i < n) p[i] = v;
}

__global__ __launch_bounds__(256)
void fill_u32(unsigned* p, unsigned v, int n) {
  int i = blockIdx.x * blockDim.x + threadIdx.x;
  if (i < n) p[i] = v;
}

__global__ __launch_bounds__(256)
void bias_fill(float* __restrict__ dst, const float* __restrict__ bias,
               int total, int mask) {
  int i = blockIdx.x * blockDim.x + threadIdx.x;
  if (i < total) dst[i] = bias[i & mask];
}

// ---------------------------------------------------------------------------
// Edge kernels. One wave per edge; H=4 heads -> 8 lanes/head for both C=128
// and C=64. Self loops are edges e >= E with src=dst=e-E.
// ---------------------------------------------------------------------------
template <int C>
__global__ __launch_bounds__(256)
void edge_score_max(const float* __restrict__ xl, const float* __restrict__ xr,
                    const float* __restrict__ att, const int* __restrict__ ei,
                    int E, int Nn, float* __restrict__ scores,
                    unsigned* __restrict__ nmax) {
  constexpr int HC  = 4 * C;
  constexpr int PL4 = HC / 32 / 4;  // float4 chunks per lane
  const int lane = threadIdx.x & 31;
  const int e = blockIdx.x * 8 + (threadIdx.x >> 5);
  if (e >= E + Nn) return;
  int s_, d_;
  if (e < E) { s_ = ei[e]; d_ = ei[E + e]; } else { s_ = d_ = e - E; }

  const float4* pl = (const float4*)(xl + (size_t)s_ * HC) + lane * PL4;
  const float4* pr = (const float4*)(xr + (size_t)d_ * HC) + lane * PL4;
  const float4* pa = (const float4*)att + lane * PL4;
  float acc = 0.f;
#pragma unroll
  for (int q = 0; q < PL4; ++q) {
    float4 a = pl[q], b = pr[q], t = pa[q];
    float v;
    v = a.x + b.x; acc += t.x * (v > 0.f ? v : 0.2f * v);
    v = a.y + b.y; acc += t.y * (v > 0.f ? v : 0.2f * v);
    v = a.z + b.z; acc += t.z * (v > 0.f ? v : 0.2f * v);
    v = a.w + b.w; acc += t.w * (v > 0.f ? v : 0.2f * v);
  }
  acc += __shfl_xor(acc, 1, 32);
  acc += __shfl_xor(acc, 2, 32);
  acc += __shfl_xor(acc, 4, 32);
  if ((lane & 7) == 0) {
    int head = lane >> 3;
    scores[(size_t)e * 4 + head] = acc;
    unsigned b = __builtin_bit_cast(unsigned, acc);
    unsigned key = b ^ ((b >> 31) ? 0xFFFFFFFFu : 0x80000000u);  // order-preserving
    atomicMax(&nmax[(size_t)d_ * 4 + head], key);
  }
}

__global__ __launch_bounds__(256)
void edge_expsum(float* __restrict__ scores, const int* __restrict__ ei,
                 int E, int Nn, const unsigned* __restrict__ nmax,
                 float* __restrict__ denom) {
  int idx = blockIdx.x * blockDim.x + threadIdx.x;
  if (idx >= (E + Nn) * 4) return;
  int e = idx >> 2, h = idx & 3;
  int d_ = (e < E) ? ei[E + e] : (e - E);
  unsigned key  = nmax[(size_t)d_ * 4 + h];
  unsigned bits = key ^ ((key >> 31) ? 0x80000000u : 0xFFFFFFFFu);
  float mx = __builtin_bit_cast(float, bits);
  float ex = __expf(scores[idx] - mx);
  scores[idx] = ex;
  atomicAdd(&denom[(size_t)d_ * 4 + h], ex);
}

template <int C>
__global__ __launch_bounds__(256)
void edge_aggregate(const float* __restrict__ xl, const float* __restrict__ scores,
                    const float* __restrict__ denom, const int* __restrict__ ei,
                    int E, int Nn, float* __restrict__ out) {
  constexpr int HC = 4 * C;
  constexpr int PL = HC / 32;
  const int lane = threadIdx.x & 31;
  const int e = blockIdx.x * 8 + (threadIdx.x >> 5);
  if (e >= E + Nn) return;
  int s_, d_;
  if (e < E) { s_ = ei[e]; d_ = ei[E + e]; } else { s_ = d_ = e - E; }
  int head = lane >> 3;
  float alpha = scores[(size_t)e * 4 + head] /
                (denom[(size_t)d_ * 4 + head] + 1e-16f);
  const float* pl = xl + (size_t)s_ * HC + lane * PL;
  float* po = out + (size_t)d_ * HC + lane * PL;
#pragma unroll
  for (int i = 0; i < PL; ++i) atomicAdd(&po[i], alpha * pl[i]);
}

// out[n] = mean_c( mean_h agg2[n,h,c] + bias2[c] )
//        = ( sum(agg2[n,:]) + 4*sum(bias2) ) / 256
__global__ __launch_bounds__(256)
void finalize(const float* __restrict__ agg2, const float* __restrict__ bias2,
              float* __restrict__ out, int Nn) {
  const int lane = threadIdx.x & 31;
  const int n = blockIdx.x * 8 + (threadIdx.x >> 5);
  if (n >= Nn) return;
  const float4* p = (const float4*)(agg2 + (size_t)n * 256);
  float s = 0.f;
#pragma unroll
  for (int i = 0; i < 2; ++i) {
    float4 v = p[lane * 2 + i];
    s += v.x + v.y + v.z + v.w;
  }
  s += 4.f * (bias2[lane] + bias2[lane + 32]);
#pragma unroll
  for (int off = 16; off > 0; off >>= 1) s += __shfl_xor(s, off, 32);
  if (lane == 0) out[n] = s * (1.0f / 256.0f);
}

// ---------------------------------------------------------------------------
// Launcher
// ---------------------------------------------------------------------------
extern "C" void kernel_launch(void* const* d_in, const int* in_sizes, int n_in,
                              void* d_out, int out_size, void* d_ws, size_t ws_size,
                              hipStream_t stream) {
  const float* x     = (const float*)d_in[0];
  const int*   ei    = (const int*)d_in[1];   // [2,E] flat, int32 (JAX x64 off)
  const float* Wl1   = (const float*)d_in[3];
  const float* bl1   = (const float*)d_in[4];
  const float* Wr1   = (const float*)d_in[5];
  const float* br1   = (const float*)d_in[6];
  const float* att1  = (const float*)d_in[7];
  const float* bias1 = (const float*)d_in[8];
  const float* Wl2   = (const float*)d_in[9];
  const float* bl2   = (const float*)d_in[10];
  const float* Wr2   = (const float*)d_in[11];
  const float* br2   = (const float*)d_in[12];
  const float* att2  = (const float*)d_in[13];
  const float* bias2 = (const float*)d_in[14];
  float* out = (float*)d_out;

  const int F = 512, HC1 = 512, HC2 = 256;
  const int Nn = in_sizes[0] / F;     // 32768 nodes
  const int E  = in_sizes[1] / 2;     // 262144 edges
  const int ET = E + Nn;              // + self loops

  // workspace bump allocator (256B aligned slots)
  char* ws = (char*)d_ws;
  size_t off = 0;
  auto alloc = [&](size_t bytes) -> char* {
    char* p = ws + off;
    off += (bytes + 255) & ~(size_t)255;
    return p;
  };
  __bf16* Abf    = (__bf16*)alloc((size_t)Nn * F * 2);     // x_bf16, later h1_bf16
  __bf16* W1lT   = (__bf16*)alloc((size_t)F * HC1 * 2);
  __bf16* W1rT   = (__bf16*)alloc((size_t)F * HC1 * 2);
  __bf16* W2lT   = (__bf16*)alloc((size_t)HC1 * HC2 * 2);
  __bf16* W2rT   = (__bf16*)alloc((size_t)HC1 * HC2 * 2);
  float*  bufP   = (float*)alloc((size_t)Nn * HC1 * 4);    // xl1; later xl2|xr2
  float*  bufQ   = (float*)alloc((size_t)Nn * HC1 * 4);    // xr1; later agg2
  float*  h1     = (float*)alloc((size_t)Nn * HC1 * 4);
  float*  scores = (float*)alloc((size_t)ET * 4 * 4);
  unsigned* nmax = (unsigned*)alloc((size_t)Nn * 4 * 4);
  float*  denom  = (float*)alloc((size_t)Nn * 4 * 4);
  (void)ws_size; (void)n_in; (void)out_size;

  float* xl1  = bufP;
  float* xr1  = bufQ;
  float* xl2  = bufP;
  float* xr2  = bufP + (size_t)Nn * HC2;
  float* agg2 = bufQ;

  const int TB = 256;
  auto cdiv = [](int a, int b) { return (a + b - 1) / b; };

  // ---- input conversions -------------------------------------------------
  cvt_f32_bf16<<<cdiv(Nn * F / 4, TB), TB, 0, stream>>>(x, Abf, Nn * F / 4);
  transpose_cvt<<<cdiv(F * HC1, TB), TB, 0, stream>>>(Wl1, W1lT, F, HC1);
  transpose_cvt<<<cdiv(F * HC1, TB), TB, 0, stream>>>(Wr1, W1rT, F, HC1);
  transpose_cvt<<<cdiv(HC1 * HC2, TB), TB, 0, stream>>>(Wl2, W2lT, HC1, HC2);
  transpose_cvt<<<cdiv(HC1 * HC2, TB), TB, 0, stream>>>(Wr2, W2rT, HC1, HC2);

  // ---- layer 1: GEMMs (WMMA, async double-buffered tiles) ----------------
  dim3 g1(HC1 / 128, Nn / 128);
  gemm_bf16_wmma<<<g1, TB, 0, stream>>>(Abf, W1lT, bl1, xl1, Nn, HC1, F);
  gemm_bf16_wmma<<<g1, TB, 0, stream>>>(Abf, W1rT, br1, xr1, Nn, HC1, F);

  // ---- layer 1: edge softmax + aggregate --------------------------------
  fill_u32<<<cdiv(Nn * 4, TB), TB, 0, stream>>>(nmax, 0u, Nn * 4);
  fill_f32<<<cdiv(Nn * 4, TB), TB, 0, stream>>>(denom, 0.f, Nn * 4);
  bias_fill<<<cdiv(Nn * HC1, TB), TB, 0, stream>>>(h1, bias1, Nn * HC1, HC1 - 1);
  edge_score_max<128><<<cdiv(ET, 8), TB, 0, stream>>>(xl1, xr1, att1, ei, E, Nn,
                                                      scores, nmax);
  edge_expsum<<<cdiv(ET * 4, TB), TB, 0, stream>>>(scores, ei, E, Nn, nmax, denom);
  edge_aggregate<128><<<cdiv(ET, 8), TB, 0, stream>>>(xl1, scores, denom, ei,
                                                      E, Nn, h1);

  // ---- layer 2: GEMMs (WMMA) --------------------------------------------
  cvt_f32_bf16<<<cdiv(Nn * HC1 / 4, TB), TB, 0, stream>>>(h1, Abf, Nn * HC1 / 4);
  dim3 g2(HC2 / 128, Nn / 128);
  gemm_bf16_wmma<<<g2, TB, 0, stream>>>(Abf, W2lT, bl2, xl2, Nn, HC2, HC1);
  gemm_bf16_wmma<<<g2, TB, 0, stream>>>(Abf, W2rT, br2, xr2, Nn, HC2, HC1);

  // ---- layer 2: edge softmax + aggregate --------------------------------
  fill_u32<<<cdiv(Nn * 4, TB), TB, 0, stream>>>(nmax, 0u, Nn * 4);
  fill_f32<<<cdiv(Nn * 4, TB), TB, 0, stream>>>(denom, 0.f, Nn * 4);
  fill_f32<<<cdiv(Nn * HC2, TB), TB, 0, stream>>>(agg2, 0.f, Nn * HC2);
  edge_score_max<64><<<cdiv(ET, 8), TB, 0, stream>>>(xl2, xr2, att2, ei, E, Nn,
                                                     scores, nmax);
  edge_expsum<<<cdiv(ET * 4, TB), TB, 0, stream>>>(scores, ei, E, Nn, nmax, denom);
  edge_aggregate<64><<<cdiv(ET, 8), TB, 0, stream>>>(xl2, scores, denom, ei,
                                                     E, Nn, agg2);

  // ---- final: mean over heads + bias + mean over features ---------------
  finalize<<<cdiv(Nn, 8), TB, 0, stream>>>(agg2, bias2, out, Nn);
}
